// HGNN_model_39041252720788
// MI455X (gfx1250) — matrix operand order repaired and checked by reading, not compiled, not run on hardware
//
#include <hip/hip_runtime.h>
#include <hip/hip_bf16.h>
#include <stdint.h>

// ---------------------------------------------------------------------------
// MI455X (gfx1250) implementation of the hetero-GraphSAGE + pair-MLP model.
// GEMMs (FLOP-dominant) run on v_wmma_f32_16x16x32_f16 (wave32), with both
// operands double-buffered through LDS: one barrier per K-step, 8 WMMAs per
// step. B panels (f16 weights) are staged with GLOBAL_LOAD_ASYNC_TO_LDS_B128
// (ASYNCcnt-tracked, VGPR-bypassing); A panels need f32->f16 cvt so they go
// through VALU + ds_store. Edge aggregation / BN are bandwidth-bound helpers.
// ---------------------------------------------------------------------------

typedef __attribute__((ext_vector_type(16))) _Float16 v16h;
typedef __attribute__((ext_vector_type(8)))  _Float16 v8h;
typedef __attribute__((ext_vector_type(4)))  _Float16 v4h;
typedef __attribute__((ext_vector_type(8)))  float    v8f;
typedef __attribute__((address_space(3)))    _Float16 lds_f16;

#define HDIM 128
#define EPSV 1e-5f
#define LDP 72   // LDS row pitch in halfs (64 data + 8 pad -> 144B, bank spread)

// -------------------------------- utility ----------------------------------

__global__ void zero_kernel(float* __restrict__ p, long long n) {
  long long i = (long long)blockIdx.x * blockDim.x + threadIdx.x;
  if (i < n) p[i] = 0.f;
}

// f32 [K,N] row-major  ->  f16 [N,K] row-major (transposed, WMMA B operand)
__global__ void wcvt_kernel(const float* __restrict__ in, _Float16* __restrict__ out,
                            int K, int N) {
  long long total = (long long)K * N;
  long long i = (long long)blockIdx.x * blockDim.x + threadIdx.x;
  if (i < total) {
    int n = (int)(i / K);
    int k = (int)(i - (long long)n * K);
    out[i] = (_Float16)in[(long long)k * N + n];
  }
}

// ------------------------------ WMMA GEMM ----------------------------------
// C[M,N] = (acc ? C : 0) + alpha * (A[M,K] @ B[K,N] + bias[N])
// A: f32 row-major (staged to LDS as f16), Bt: f16 [N,K] (pre-transposed,
// async-staged to LDS). Block = 128 threads = 4 waves. Block tile 64x64,
// K-step 64, double-buffered. Each wave: 16(M) x 64(N) strip = 8 WMMAs/step.

__device__ __forceinline__ void stage_panels(const float* __restrict__ A,
                                             const _Float16* __restrict__ Bt,
                                             int M, int K, int m_blk, int n_blk,
                                             int k0, int tid,
                                             _Float16 (*__restrict__ As)[LDP],
                                             _Float16 (*__restrict__ Bs)[LDP]) {
  const int a_row = tid >> 1;                      // 0..63
  const int a_c0  = (tid & 1) << 5;                // 0 or 32
  const int a_gr  = m_blk + a_row;
  const bool full = (m_blk + 64 <= M) && (k0 + 64 <= K);   // block-uniform
  if (full) {
    const float* ap = A + (size_t)a_gr * K + k0 + a_c0;
#pragma unroll
    for (int i = 0; i < 8; ++i) {
      float4 v = *(const float4*)(ap + (i << 2));
      v4h h;
      h[0] = (_Float16)v.x; h[1] = (_Float16)v.y;
      h[2] = (_Float16)v.z; h[3] = (_Float16)v.w;
      *(v4h*)&As[a_row][a_c0 + (i << 2)] = h;
    }
    // B panel: async global->LDS copy, tracked on ASYNCcnt (no VGPR staging)
#pragma unroll
    for (int i = 0; i < 4; ++i) {
      int c  = (tid << 2) + i;                     // 0..511 chunks of 8 halfs
      int n  = c >> 3;
      int ko = (c & 7) << 3;
      const _Float16* src = Bt + (size_t)(n_blk + n) * K + k0 + ko;
      unsigned int dst = (unsigned int)(uintptr_t)(lds_f16*)&Bs[n][ko];
      asm volatile("global_load_async_to_lds_b128 %0, %1, off"
                   :: "v"(dst), "v"(src) : "memory");
    }
    if (k0 + 64 < K)                               // uniform; next-next panel
      __builtin_prefetch(ap + 64, 0, 3);
  } else {
#pragma unroll
    for (int i = 0; i < 8; ++i) {
      int kc = a_c0 + (i << 2);
      int gk = k0 + kc;
      float4 v = make_float4(0.f, 0.f, 0.f, 0.f);
      if (a_gr < M && gk < K) v = *(const float4*)(A + (size_t)a_gr * K + gk);
      v4h h;
      h[0] = (_Float16)v.x; h[1] = (_Float16)v.y;
      h[2] = (_Float16)v.z; h[3] = (_Float16)v.w;
      *(v4h*)&As[a_row][kc] = h;
    }
#pragma unroll
    for (int i = 0; i < 4; ++i) {
      int c  = (tid << 2) + i;
      int n  = c >> 3;
      int ko = (c & 7) << 3;
      int gk = k0 + ko;
      v8h v = {};
      if (gk < K) v = *(const v8h*)(Bt + (size_t)(n_blk + n) * K + gk);
      *(v8h*)&Bs[n][ko] = v;
    }
  }
}

__device__ __forceinline__ void compute_step(const _Float16 (*__restrict__ As)[LDP],
                                             const _Float16 (*__restrict__ Bs)[LDP],
                                             v8f accv[4], int am, int akb,
                                             int bn_l, int bkh) {
#pragma unroll
  for (int s = 0; s < 2; ++s) {
    const int kb = s << 5;
    v8h alo = *(const v8h*)&As[am][kb + akb];       // ds_load_b128
    v8h ahi = *(const v8h*)&As[am][kb + akb + 8];
    v16h af;
#pragma unroll
    for (int i = 0; i < 8; ++i) { af[i] = alo[i]; af[i + 8] = ahi[i]; }
    v16h bf[4];
#pragma unroll
    for (int j = 0; j < 4; ++j) {                   // issue all B loads first
      const _Float16* bp = &Bs[(j << 4) + bn_l][kb + bkh];
      v8h blo = *(const v8h*)bp;
      v8h bhi = *(const v8h*)(bp + 8);
#pragma unroll
      for (int i = 0; i < 8; ++i) { bf[j][i] = blo[i]; bf[j][i + 8] = bhi[i]; }
    }
#pragma unroll
    for (int j = 0; j < 4; ++j)
      accv[j] = __builtin_amdgcn_wmma_f32_16x16x32_f16(
          false, af, false, bf[j], (short)0, accv[j], false, false);
  }
}

__launch_bounds__(128)
__global__ void gemm_wmma_kernel(const float* __restrict__ A,
                                 const _Float16* __restrict__ Bt,
                                 const float* __restrict__ bias,
                                 float* __restrict__ C,
                                 int M, int N, int K, float alpha, int acc) {
  __shared__ __align__(16) _Float16 As[2][64][LDP];
  __shared__ __align__(16) _Float16 Bs[2][64][LDP];

  const int tid  = threadIdx.x;
  const int lane = tid & 31;
  const int wave = tid >> 5;                       // 0..3 : M sub-tile
  const int m_blk = blockIdx.y << 6;
  const int n_blk = blockIdx.x << 6;

  // A fragment: lanes 0-15 -> K {0..15}; lanes 16-31 -> K {16..31}
  const int am  = (wave << 4) + (lane & 15);
  const int akb = (lane >> 4) << 4;                // 0 / 16
  // B fragment: n = lane&15, K = (lane>>4)*16 + e  (contiguous)
  const int bn_l = lane & 15;
  const int bkh  = (lane >> 4) << 4;               // 0 / 16

  v8f accv[4];
  {
    v8f z = {};
#pragma unroll
    for (int j = 0; j < 4; ++j) accv[j] = z;
  }

  // prologue: stage first panel into buffer 0
  stage_panels(A, Bt, M, K, m_blk, n_blk, 0, tid, As[0], Bs[0]);

  int buf = 0;
  for (int k0 = 0; k0 < K; k0 += 64) {
    // my async B-copies must land in LDS before I signal the barrier
    asm volatile("s_wait_asynccnt 0" ::: "memory");
    __syncthreads();   // panel[buf] staged; compute on buf^1 (prev iter) done
    if (k0 + 64 < K)
      stage_panels(A, Bt, M, K, m_blk, n_blk, k0 + 64, tid, As[buf ^ 1], Bs[buf ^ 1]);
    compute_step(As[buf], Bs[buf], accv, am, akb, bn_l, bkh);
    buf ^= 1;
  }

  // ---- epilogue: alpha*(acc + bias), optional accumulate into C ----
  const int mrow0 = m_blk + (wave << 4) + ((lane >> 4) << 3);
  if (m_blk + 64 <= M) {                           // uniform fast path
#pragma unroll
    for (int j = 0; j < 4; ++j) {
      int n = n_blk + (j << 4) + bn_l;
      float bv = bias ? bias[n] : 0.f;
#pragma unroll
      for (int r = 0; r < 8; ++r) {
        size_t off = (size_t)(mrow0 + r) * N + n;
        float res = alpha * (accv[j][r] + bv);
        if (acc) res += C[off];
        C[off] = res;
      }
    }
  } else {
#pragma unroll
    for (int j = 0; j < 4; ++j) {
      int n = n_blk + (j << 4) + bn_l;
      float bv = bias ? bias[n] : 0.f;
#pragma unroll
      for (int r = 0; r < 8; ++r) {
        int m = mrow0 + r;
        if (m < M) {
          size_t off = (size_t)m * N + n;
          float res = alpha * (accv[j][r] + bv);
          if (acc) res += C[off];
          C[off] = res;
        }
      }
    }
  }
}

// --------------------------- graph aggregation -----------------------------

// sum[dst] += X[src] over edges; cnt[dst] += 1.  One (edge, 4-float chunk)
// per thread; consecutive threads read consecutive 16B of the same row.
__global__ void scatter_add_kernel(const float* __restrict__ X,
                                   const int* __restrict__ ei, int E,
                                   float* __restrict__ sum, float* __restrict__ cnt) {
  long long i = (long long)blockIdx.x * blockDim.x + threadIdx.x;
  long long total = (long long)E * 32;
  if (i >= total) return;
  int e = (int)(i >> 5);
  int c = (int)(i & 31);
  int s = ei[e];
  int d = ei[E + e];
  float4 v = *(const float4*)(X + (size_t)s * HDIM + (c << 2));
  float* sp = sum + (size_t)d * HDIM + (c << 2);
  atomicAdd(sp + 0, v.x);
  atomicAdd(sp + 1, v.y);
  atomicAdd(sp + 2, v.z);
  atomicAdd(sp + 3, v.w);
  if (c == 0) atomicAdd(cnt + d, 1.0f);
}

__global__ void mean_div_kernel(float* __restrict__ sum, const float* __restrict__ cnt,
                                long long n) {                  // n = nodes*128
  long long i = (long long)blockIdx.x * blockDim.x + threadIdx.x;
  if (i < n) sum[i] /= fmaxf(cnt[i >> 7], 1.0f);
}

// ev[p] = concat(X[src[p]], X[dst[p]])   (float4 chunks)
__global__ void pair_gather_kernel(const float* __restrict__ X,
                                   const int* __restrict__ ei, int P,
                                   float* __restrict__ ev) {
  long long i = (long long)blockIdx.x * blockDim.x + threadIdx.x;
  long long total = (long long)P * 64;
  if (i >= total) return;
  int p = (int)(i >> 6);
  int c = (int)(i & 63);
  int node = (c < 32) ? ei[p] : ei[P + p];
  float4 v = *(const float4*)(X + (size_t)node * HDIM + ((c & 31) << 2));
  *(float4*)(ev + (size_t)p * 256 + (c << 2)) = v;
}

// ------------------------------- batch norm --------------------------------

__global__ void bn_stats_kernel(const float* __restrict__ h, int P, int C,
                                float* __restrict__ sum, float* __restrict__ sumsq) {
  int col = threadIdx.x;                       // blockDim.x == C
  int rpb = (P + gridDim.x - 1) / gridDim.x;
  int r0 = blockIdx.x * rpb;
  int r1 = min(P, r0 + rpb);
  float s = 0.f, s2 = 0.f;
  for (int r = r0; r < r1; ++r) {
    float v = h[(size_t)r * C + col];
    s += v; s2 += v * v;
  }
  atomicAdd(sum + col, s);
  atomicAdd(sumsq + col, s2);
}

__global__ void bn_finalize_kernel(const float* __restrict__ sum,
                                   const float* __restrict__ sumsq,
                                   const float* __restrict__ g,
                                   const float* __restrict__ b,
                                   int C, float invP,
                                   float* __restrict__ scale, float* __restrict__ shift) {
  int c = blockIdx.x * blockDim.x + threadIdx.x;
  if (c < C) {
    float mu  = sum[c] * invP;
    float var = sumsq[c] * invP - mu * mu;
    float sc  = g[c] * rsqrtf(var + EPSV);
    scale[c] = sc;
    shift[c] = b[c] - mu * sc;
  }
}

__global__ void bn_relu_kernel(float* __restrict__ h, long long total, int cmask,
                               const float* __restrict__ scale,
                               const float* __restrict__ shift) {
  long long i = (long long)blockIdx.x * blockDim.x + threadIdx.x;
  if (i < total) {
    int c = (int)(i & cmask);                  // C is a power of two
    h[i] = fmaxf(h[i] * scale[c] + shift[c], 0.f);
  }
}

// out[p] = h3[p,0:64] . W4 + b4
__global__ void final_dot_kernel(const float* __restrict__ h3,
                                 const float* __restrict__ W4,
                                 const float* __restrict__ b4,
                                 float* __restrict__ out, int P) {
  __shared__ float w[64];
  if (threadIdx.x < 64) w[threadIdx.x] = W4[threadIdx.x];
  __syncthreads();
  int p = blockIdx.x * blockDim.x + threadIdx.x;
  if (p < P) {
    float s = b4[0];
#pragma unroll
    for (int i = 0; i < 64; ++i) s += h3[(size_t)p * 64 + i] * w[i];
    out[p] = s;
  }
}

// ------------------------------- launcher ----------------------------------

extern "C" void kernel_launch(void* const* d_in, const int* in_sizes, int n_in,
                              void* d_out, int out_size, void* d_ws, size_t ws_size,
                              hipStream_t stream) {
  (void)n_in; (void)out_size; (void)ws_size;

  const float* x_oer  = (const float*)d_in[0];
  const float* x_con  = (const float*)d_in[1];
  const float* x_cls  = (const float*)d_in[2];
  const float* W_oer  = (const float*)d_in[3];
  const float* b_oer  = (const float*)d_in[4];
  const float* W_con  = (const float*)d_in[5];
  const float* b_con  = (const float*)d_in[6];
  const float* W_cls  = (const float*)d_in[7];
  const float* b_cls  = (const float*)d_in[8];
  const float* conv_Wl = (const float*)d_in[9];
  const float* conv_bl = (const float*)d_in[10];
  const float* conv_Wr = (const float*)d_in[11];
  const float* mlp_W[5] = {(const float*)d_in[12], (const float*)d_in[16],
                           (const float*)d_in[20], (const float*)d_in[24],
                           (const float*)d_in[28]};
  const float* mlp_b[5] = {(const float*)d_in[13], (const float*)d_in[17],
                           (const float*)d_in[21], (const float*)d_in[25],
                           (const float*)d_in[29]};
  const float* bn_g[4] = {(const float*)d_in[14], (const float*)d_in[18],
                          (const float*)d_in[22], (const float*)d_in[26]};
  const float* bn_b[4] = {(const float*)d_in[15], (const float*)d_in[19],
                          (const float*)d_in[23], (const float*)d_in[27]};
  const int* ei_sr   = (const int*)d_in[30];
  const int* ei_ep   = (const int*)d_in[31];
  const int* ei_cov  = (const int*)d_in[32];
  const int* ei_bel  = (const int*)d_in[33];
  const int* ei_rcov = (const int*)d_in[34];
  const int* ei_rbel = (const int*)d_in[35];
  const int E_sr  = in_sizes[30] / 2, E_ep  = in_sizes[31] / 2,
            E_cov = in_sizes[32] / 2, E_bel = in_sizes[33] / 2,
            E_rcov = in_sizes[34] / 2, E_rbel = in_sizes[35] / 2;

  const int NO = 200000, NC = 100000, NK = 10000, H = HDIM;
  const int P  = E_sr;

  // ---- workspace carve (256B aligned bump allocator) ----
  char* wsp = (char*)d_ws;
  auto alloc = [&](size_t bytes) -> char* {
    char* p = wsp;
    wsp += (bytes + 255) & ~(size_t)255;
    return p;
  };
  float* xO[2] = {(float*)alloc((size_t)NO * H * 4), (float*)alloc((size_t)NO * H * 4)};
  float* xC[2] = {(float*)alloc((size_t)NC * H * 4), (float*)alloc((size_t)NC * H * 4)};
  float* xK[2] = {(float*)alloc((size_t)NK * H * 4), (float*)alloc((size_t)NK * H * 4)};
  float* meanb = (float*)alloc((size_t)NO * H * 4);
  float* cntb  = (float*)alloc((size_t)NO * 4);
  _Float16* WtO = (_Float16*)alloc(128 * 128 * 2);
  _Float16* WtC = (_Float16*)alloc(64 * 128 * 2);
  _Float16* WtK = (_Float16*)alloc(32 * 128 * 2);
  _Float16* WtCL = (_Float16*)alloc((size_t)10 * H * H * 2);
  _Float16* WtCR = (_Float16*)alloc((size_t)10 * H * H * 2);
  _Float16* WtM[4] = {(_Float16*)alloc(256 * 512 * 2), (_Float16*)alloc(512 * 256 * 2),
                      (_Float16*)alloc(256 * 128 * 2), (_Float16*)alloc(128 * 64 * 2)};
  float* big0 = (float*)alloc((size_t)P * 256 * 4);   // ev, h1, h3
  float* big1 = (float*)alloc((size_t)P * 512 * 4);   // h0, h2
  float* bnsum   = (float*)alloc(512 * 2 * 4);        // sum | sumsq
  float* bnscale = (float*)alloc(512 * 4);
  float* bnshift = (float*)alloc(512 * 4);

  // ---- launch helpers ----
  auto zerol = [&](float* p, long long n) {
    zero_kernel<<<dim3((unsigned)((n + 255) / 256)), dim3(256), 0, stream>>>(p, n);
  };
  auto wcvt = [&](const float* in, _Float16* out, int K, int N) {
    long long t = (long long)K * N;
    wcvt_kernel<<<dim3((unsigned)((t + 255) / 256)), dim3(256), 0, stream>>>(in, out, K, N);
  };
  auto gemm = [&](const float* A, const _Float16* Bt, const float* bias, float* C,
                  int M, int N, int K, float alpha, int acc) {
    dim3 g((unsigned)(N / 64), (unsigned)((M + 63) / 64));
    gemm_wmma_kernel<<<g, dim3(128), 0, stream>>>(A, Bt, bias, C, M, N, K, alpha, acc);
  };
  auto scat = [&](const float* X, const int* ei, int E) {
    long long t = (long long)E * 32;
    scatter_add_kernel<<<dim3((unsigned)((t + 255) / 256)), dim3(256), 0, stream>>>(
        X, ei, E, meanb, cntb);
  };
  auto mdiv = [&](int nodes) {
    long long t = (long long)nodes * H;
    mean_div_kernel<<<dim3((unsigned)((t + 255) / 256)), dim3(256), 0, stream>>>(meanb, cntb, t);
  };

  // ---- weight pre-transpose / f16 conversion ----
  wcvt(W_oer, WtO, 128, 128);
  wcvt(W_con, WtC, 64, 128);
  wcvt(W_cls, WtK, 32, 128);
  for (int l = 0; l < 2; ++l)
    for (int t = 0; t < 5; ++t) {
      size_t o = (size_t)(l * 5 + t) * H * H;
      wcvt(conv_Wl + o, WtCL + o, H, H);
      wcvt(conv_Wr + o, WtCR + o, H, H);
    }
  wcvt(mlp_W[0], WtM[0], 256, 512);
  wcvt(mlp_W[1], WtM[1], 512, 256);
  wcvt(mlp_W[2], WtM[2], 256, 128);
  wcvt(mlp_W[3], WtM[3], 128, 64);

  // ---- per-type input projections ----
  gemm(x_oer, WtO, b_oer, xO[0], NO, 128, 128, 1.f, 0);
  gemm(x_con, WtC, b_con, xC[0], NC, 128, 64, 1.f, 0);
  gemm(x_cls, WtK, b_cls, xK[0], NK, 128, 32, 1.f, 0);

  // ---- 2 HeteroConv layers ----
  int cur = 0, nxt = 1;
  for (int l = 0; l < 2; ++l) {
    auto Wl = [&](int t) { return WtCL + (size_t)(l * 5 + t) * H * H; };
    auto Wr = [&](int t) { return WtCR + (size_t)(l * 5 + t) * H * H; };
    auto bl = [&](int t) { return conv_bl + (size_t)(l * 5 + t) * H; };

    // OER dst: 0.5*(sage(before_ep) + sage(rev_covers))
    zerol(meanb, (long long)NO * H); zerol(cntb, NO);
    scat(xO[cur], ei_ep, E_ep); mdiv(NO);
    gemm(meanb,   Wl(0), bl(0),  xO[nxt], NO, 128, 128, 0.5f, 0);
    gemm(xO[cur], Wr(0), nullptr, xO[nxt], NO, 128, 128, 0.5f, 1);
    zerol(meanb, (long long)NO * H); zerol(cntb, NO);
    scat(xC[cur], ei_rcov, E_rcov); mdiv(NO);
    gemm(meanb,   Wl(3), bl(3),  xO[nxt], NO, 128, 128, 0.5f, 1);
    gemm(xO[cur], Wr(3), nullptr, xO[nxt], NO, 128, 128, 0.5f, 1);

    // CON dst: 0.5*(sage(covers) + sage(rev_belongs))
    zerol(meanb, (long long)NC * H); zerol(cntb, NC);
    scat(xO[cur], ei_cov, E_cov); mdiv(NC);
    gemm(meanb,   Wl(1), bl(1),  xC[nxt], NC, 128, 128, 0.5f, 0);
    gemm(xC[cur], Wr(1), nullptr, xC[nxt], NC, 128, 128, 0.5f, 1);
    zerol(meanb, (long long)NC * H); zerol(cntb, NC);
    scat(xK[cur], ei_rbel, E_rbel); mdiv(NC);
    gemm(meanb,   Wl(4), bl(4),  xC[nxt], NC, 128, 128, 0.5f, 1);
    gemm(xC[cur], Wr(4), nullptr, xC[nxt], NC, 128, 128, 0.5f, 1);

    // CLS dst: sage(belongs)
    zerol(meanb, (long long)NK * H); zerol(cntb, NK);
    scat(xC[cur], ei_bel, E_bel); mdiv(NK);
    gemm(meanb,   Wl(2), bl(2),  xK[nxt], NK, 128, 128, 1.f, 0);
    gemm(xK[cur], Wr(2), nullptr, xK[nxt], NK, 128, 128, 1.f, 1);

    cur ^= 1; nxt ^= 1;
  }

  // ---- pair features + MLP with batch-norm/relu ----
  {
    long long t = (long long)P * 64;
    pair_gather_kernel<<<dim3((unsigned)((t + 255) / 256)), dim3(256), 0, stream>>>(
        xO[cur], ei_sr, P, big0);
  }
  const int Kin[4]  = {256, 512, 256, 128};
  const int Nout[4] = {512, 256, 128, 64};
  float* bufs[2] = {big0, big1};
  int src = 0;
  for (int i = 0; i < 4; ++i) {
    int dst = src ^ 1;
    gemm(bufs[src], WtM[i], mlp_b[i], bufs[dst], P, Nout[i], Kin[i], 1.f, 0);
    zerol(bnsum, 2 * Nout[i]);
    bn_stats_kernel<<<dim3(512), dim3(Nout[i]), 0, stream>>>(bufs[dst], P, Nout[i],
                                                             bnsum, bnsum + Nout[i]);
    bn_finalize_kernel<<<dim3(1), dim3(Nout[i]), 0, stream>>>(
        bnsum, bnsum + Nout[i], bn_g[i], bn_b[i], Nout[i], 1.f / (float)P,
        bnscale, bnshift);
    long long tot = (long long)P * Nout[i];
    bn_relu_kernel<<<dim3((unsigned)((tot + 255) / 256)), dim3(256), 0, stream>>>(
        bufs[dst], tot, Nout[i] - 1, bnscale, bnshift);
    src = dst;
  }
  final_dot_kernel<<<dim3((unsigned)((P + 255) / 256)), dim3(256), 0, stream>>>(
      bufs[src], mlp_W[4], mlp_b[4], (float*)d_out, P);
}
